// ScaledDotProduct_62706522522246
// MI455X (gfx1250) — matrix operand order repaired and checked by reading, not compiled, not run on hardware
//
#include <hip/hip_runtime.h>
#include <hip/hip_bf16.h>

typedef __attribute__((ext_vector_type(16))) __bf16 v16bf;
typedef __attribute__((ext_vector_type(8)))  float  v8f;

// Problem geometry (fixed by the reference).
constexpr int BB = 4;
constexpr int HH = 16;
constexpr int SS = 2048;
constexpr int DD = 128;

constexpr int QT     = 16;            // q rows per workgroup
constexpr int NWAVE  = 8;
constexpr int BLOCK  = NWAVE * 32;    // 256 threads, wave32
constexpr int LDS_BYTES = QT * SS * 4; // 128 KB scores/probs buffer

__global__ __launch_bounds__(BLOCK)
void ScaledDotProduct_62706522522246_kernel(const float* __restrict__ q,
                                            const float* __restrict__ k,
                                            const float* __restrict__ v,
                                            const int*   __restrict__ mask,
                                            float* __restrict__ out,
                                            float* __restrict__ attn)
{
    extern __shared__ float sS[];     // [QT][SS] fp32 scores -> probs

    const int tid   = threadIdx.x;
    const int wave  = tid >> 5;
    const int lane  = tid & 31;
    const int lhalf = lane >> 4;      // 0: lanes 0-15, 1: lanes 16-31
    const int l16   = lane & 15;

    const int qblk = blockIdx.x;      // 0 .. S/QT-1
    const int h    = blockIdx.y;
    const int b    = blockIdx.z;
    const size_t bh = (size_t)b * HH + h;

    const float* Q = q + (bh * SS + (size_t)qblk * QT) * DD;
    const float* K = k + bh * SS * DD;
    const float* V = v + bh * SS * DD;
    const int*   M = mask + ((size_t)b * SS + (size_t)qblk * QT) * SS;
    float* Oout = out  + (bh * SS + (size_t)qblk * QT) * DD;
    float* Aout = attn + (bh * SS + (size_t)qblk * QT) * SS;

    // ---------------- Q A-fragments: 16x128 -> 4 fragments (K=32 each) ------
    // 16-bit A layout: lane(l16) = row M; lhalf=0 holds K {0..7,16..23},
    // lhalf=1 holds K {8..15,24..31} (per CDNA5 ISA).
    v16bf aq[4];
    {
        const int m   = l16;
        const int kb8 = lhalf * 8;
        #pragma unroll
        for (int c = 0; c < 4; ++c) {
            const float* qrow = Q + (size_t)m * DD + c * 32;
            #pragma unroll
            for (int hh = 0; hh < 8; ++hh)
                aq[c][hh] = (__bf16)qrow[kb8 + hh];
            #pragma unroll
            for (int hh = 0; hh < 8; ++hh)
                aq[c][8 + hh] = (__bf16)qrow[16 + kb8 + hh];
        }
    }

    // ---------------- GEMM1: raw scores = Q . K^T into LDS ------------------
    // Each wave owns n-blocks nb = wave + 8*j (16 iterations covers 128 blocks).
    for (int j = 0; j < SS / 16 / NWAVE; ++j) {
        const int nb = wave + j * NWAVE;
        v8f acc = {};
        #pragma unroll
        for (int c = 0; c < 4; ++c) {
            // B fragment (32x16): lane holds K-row n = nb*16+l16,
            // elements d = c*32 + lhalf*8 + {0..7,16..23} (same K order as A).
            v16bf bk;
            const float* krow = K + (size_t)(nb * 16 + l16) * DD + c * 32;
            #pragma unroll
            for (int hh = 0; hh < 16; ++hh)
                bk[hh] = (__bf16)krow[(hh & 8) * 2 + lhalf * 8 + (hh & 7)];
            acc = __builtin_amdgcn_wmma_f32_16x16x32_bf16(
                false, aq[c], false, bk, (short)0, acc, false, false);
        }
        // C/D layout: VGPR r -> row r + 8*lhalf, col = lane&15.
        #pragma unroll
        for (int r = 0; r < 8; ++r)
            sS[(r + 8 * lhalf) * SS + nb * 16 + l16] = acc[r];
    }
    __syncthreads();

    // ---------------- masked, scaled softmax per row ------------------------
    {
        const float scale = 0.08838834764831845f;  // 1/sqrt(128)
        const int row = tid >> 4;                  // 16 threads per row
        const int sub = tid & 15;
        float* srow = sS + row * SS;
        const int* mrow = M + (size_t)row * SS;

        float mx = -3.0e38f;
        for (int c = sub; c < SS; c += 16) {
            float s = (mrow[c] == 0) ? -9.0e15f : srow[c] * scale;
            srow[c] = s;
            mx = fmaxf(mx, s);
        }
        #pragma unroll
        for (int off = 1; off < 16; off <<= 1)
            mx = fmaxf(mx, __shfl_xor(mx, off, 32));

        float sum = 0.0f;
        for (int c = sub; c < SS; c += 16) {
            float e = __expf(srow[c] - mx);
            srow[c] = e;
            sum += e;
        }
        #pragma unroll
        for (int off = 1; off < 16; off <<= 1)
            sum += __shfl_xor(sum, off, 32);

        const float inv = 1.0f / sum;
        for (int c = sub; c < SS; c += 16)
            srow[c] *= inv;
    }
    __syncthreads();

    // ---------------- write attn to global (coalesced) ----------------------
    for (int i = tid; i < QT * SS; i += BLOCK)
        Aout[i] = sS[i];

    // ---------------- GEMM2: out = P . V ------------------------------------
    {
        const int nb  = wave;         // out column block: 8 waves x 16 = 128
        const int m   = l16;
        const int kb8 = lhalf * 8;
        v8f acc = {};
        for (int kb = 0; kb < SS; kb += 32) {
            // A fragment from probs in LDS (bf16 convert, interleaved K layout).
            v16bf ap;
            const float* prow = sS + m * SS + kb;
            #pragma unroll
            for (int hh = 0; hh < 8; ++hh)
                ap[hh] = (__bf16)prow[kb8 + hh];
            #pragma unroll
            for (int hh = 0; hh < 8; ++hh)
                ap[8 + hh] = (__bf16)prow[16 + kb8 + hh];
            // B fragment from V: lane col n = nb*16+l16, rows kb + lhalf*8 + {0..7,16..23}.
            // For fixed hh, lanes 0-15 read 16 consecutive floats of one V row.
            v16bf bv;
            const float* vcol = V + (size_t)kb * DD + nb * 16 + l16;
            #pragma unroll
            for (int hh = 0; hh < 16; ++hh)
                bv[hh] = (__bf16)vcol[((hh & 8) * 2 + lhalf * 8 + (hh & 7)) * DD];
            acc = __builtin_amdgcn_wmma_f32_16x16x32_bf16(
                false, ap, false, bv, (short)0, acc, false, false);
        }
        #pragma unroll
        for (int r = 0; r < 8; ++r)
            Oout[(size_t)(r + 8 * lhalf) * DD + nb * 16 + l16] = acc[r];
    }
}

extern "C" void kernel_launch(void* const* d_in, const int* in_sizes, int n_in,
                              void* d_out, int out_size, void* d_ws, size_t ws_size,
                              hipStream_t stream) {
    (void)in_sizes; (void)n_in; (void)d_ws; (void)ws_size; (void)out_size;

    const float* q    = (const float*)d_in[0];
    const float* k    = (const float*)d_in[1];
    const float* v    = (const float*)d_in[2];
    const int*   mask = (const int*)d_in[3];

    float* out  = (float*)d_out;                                   // [B,H,S,D]
    float* attn = out + (size_t)BB * HH * SS * DD;                 // [B,H,S,S]

    dim3 grid(SS / QT, HH, BB);   // 128 x 16 x 4 = 8192 workgroups
    dim3 block(BLOCK);
    ScaledDotProduct_62706522522246_kernel<<<grid, block, LDS_BYTES, stream>>>(
        q, k, v, mask, out, attn);
}
